// MambaClassifier_63118839382422
// MI455X (gfx1250) — matrix operand order repaired and checked by reading, not compiled, not run on hardware
//
#include <hip/hip_runtime.h>
#include <hip/hip_bf16.h>
#include <math.h>

typedef __bf16 bf16;
typedef __bf16 bf16x8 __attribute__((ext_vector_type(8)));
typedef __bf16 v16bf  __attribute__((ext_vector_type(16)));
typedef float  f32x8  __attribute__((ext_vector_type(8)));

#define EPI_BIAS_F32   0
#define EPI_SIGMOID    1
#define EPI_GATE_BF16  2
#define EPI_SKIP_F32   3

// ---- CDNA5 async DMA global->LDS (ASYNCcnt-tracked), 16B per lane ----------
__device__ __forceinline__ void async_copy16(bf16* ldsDst, const bf16* gsrc) {
    unsigned l = (unsigned)(unsigned long long)ldsDst;   // low 32b of generic = LDS offset
    asm volatile("global_load_async_to_lds_b128 %0, %1, off"
                 :: "v"(l), "v"(gsrc) : "memory");
}
#define WAIT_ASYNC_4() asm volatile("s_wait_asynccnt 0x4" ::: "memory")
#define WAIT_ASYNC_0() asm volatile("s_wait_asynccnt 0x0" ::: "memory")

// ---------------- WMMA fragment loaders (CDNA5 16x16x32 bf16 layouts) -------
// A (16x32 MxK): lanes 0-15 row m=l, k in {g*8..g*8+7, 16+g*8..23+g*8}, g=l>>4
__device__ __forceinline__ v16bf frag_a(const bf16* lds, int rowBase, int stride) {
    int lane = threadIdx.x & 31;
    int m = rowBase + (lane & 15);
    int g = lane >> 4;
    union { v16bf v; bf16x8 h[2]; } u;
    u.h[0] = *(const bf16x8*)(lds + m * stride + g * 8);
    u.h[1] = *(const bf16x8*)(lds + m * stride + 16 + g * 8);
    return u.v;
}
// B (32x16 KxN) from LDS stored as Wt[n][k]: lane group g holds k in {g*8..g*8+7, 16+g*8..23+g*8}
__device__ __forceinline__ v16bf frag_b(const bf16* lds, int colBase, int stride) {
    int lane = threadIdx.x & 31;
    int n = colBase + (lane & 15);
    int g = lane >> 4;
    union { v16bf v; bf16x8 h[2]; } u;
    const bf16* p = lds + n * stride + g * 8;
    u.h[0] = *(const bf16x8*)(p);
    u.h[1] = *(const bf16x8*)(p + 16);
    return u.v;
}

// ---------------- Generic NT bf16 GEMM: C[M,N] = A[M,K]*W[N,K]^T ------------
// Double-buffered LDS, async global->LDS DMA overlapped with WMMA.
__device__ __forceinline__ void gemm_issue(const bf16* __restrict__ A, const bf16* __restrict__ W,
                                           bf16* bufA, bf16* bufW,
                                           int tileM, int tileN, int K, int k0, int tidx) {
    #pragma unroll
    for (int i = 0; i < 2; i++) {
        int cid = tidx + i * 128;               // 256 chunks of 8 bf16 (16B)
        int r = cid >> 2, off = (cid & 3) * 8;
        async_copy16(bufA + r * 32 + off, A + (size_t)(tileM + r) * K + k0 + off);
        async_copy16(bufW + r * 32 + off, W + (size_t)(tileN + r) * K + k0 + off);
    }
}

__global__ __launch_bounds__(128)
void gemm_nt_bf16(const bf16* __restrict__ A, const bf16* __restrict__ W,
                  const float* __restrict__ bias, const float* __restrict__ aux,
                  float* __restrict__ outF, bf16* __restrict__ outB,
                  int M, int N, int K, int epi)
{
    __shared__ __align__(16) bf16 lA[2][64 * 32];
    __shared__ __align__(16) bf16 lW[2][64 * 32];
    int tidx = threadIdx.x;
    int wid  = tidx >> 5;
    int tileN = blockIdx.x * 64;
    int tileM = blockIdx.y * 64;

    f32x8 acc[4];
    #pragma unroll
    for (int nt = 0; nt < 4; nt++)
        #pragma unroll
        for (int r = 0; r < 8; r++) acc[nt][r] = 0.f;

    gemm_issue(A, W, lA[0], lW[0], tileM, tileN, K, 0, tidx);
    int cur = 0;
    for (int k0 = 0; k0 < K; k0 += 32) {
        if (k0 + 32 < K) {
            gemm_issue(A, W, lA[cur ^ 1], lW[cur ^ 1], tileM, tileN, K, k0 + 32, tidx);
            WAIT_ASYNC_4();                    // current tile's 4 DMAs done; next in flight
        } else {
            WAIT_ASYNC_0();
        }
        __syncthreads();
        v16bf af = frag_a(lA[cur], wid * 16, 32);
        #pragma unroll
        for (int nt = 0; nt < 4; nt++) {
            v16bf bfw = frag_b(lW[cur], nt * 16, 32);
            acc[nt] = __builtin_amdgcn_wmma_f32_16x16x32_bf16(
                false, af, false, bfw, (short)0, acc[nt], false, false);
        }
        __syncthreads();
        cur ^= 1;
    }

    int lane = tidx & 31, g = lane >> 4, nl = lane & 15;
    #pragma unroll
    for (int nt = 0; nt < 4; nt++) {
        int n = tileN + nt * 16 + nl;
        float bv = bias ? bias[n] : 0.f;
        #pragma unroll
        for (int r = 0; r < 8; r++) {
            int m = tileM + wid * 16 + g * 8 + r;
            size_t idx = (size_t)m * N + n;
            float v = acc[nt][r] + bv;
            if (epi == EPI_SIGMOID)        outF[idx] = 1.f / (1.f + __expf(-v));
            else if (epi == EPI_GATE_BF16) outB[idx] = (bf16)(v * aux[idx]);
            else if (epi == EPI_SKIP_F32)  outF[idx] = v + aux[idx];
            else                           outF[idx] = v;
        }
    }
}

// ---------------- 3x3 conv (Cin=64, pad=1), implicit GEMM, fully async ------
// X is a pre-converted bf16 NHWC activation copy; halo rows DMA from zero page
// (pointer select, not branch -> uniform per-wave ASYNCcnt).
__device__ __forceinline__ void conv_issue(const bf16* __restrict__ X, const bf16* __restrict__ W,
                                           const bf16* __restrict__ zp,
                                           bf16* bufA, bf16* bufW,
                                           int tileM, int tileN, int k0,
                                           int H, int Wd, int HW, int tidx) {
    int tap = k0 >> 6, c0 = k0 & 63;
    int dh = tap / 3 - 1, dw = (tap - 3 * (tap / 3)) - 1;
    #pragma unroll
    for (int i = 0; i < 2; i++) {
        int cid = tidx + i * 128;
        int r = cid >> 2, off = (cid & 3) * 8;
        int m = tileM + r;
        int b = m / HW; int rem = m - b * HW;
        int h = rem / Wd; int w = rem - h * Wd;
        int ih = h + dh, iw = w + dw;
        bool in = (ih >= 0) && (ih < H) && (iw >= 0) && (iw < Wd);
        const bf16* src = in ? (X + (((size_t)b * H + ih) * Wd + iw) * 64 + c0 + off) : zp;
        async_copy16(bufA + r * 32 + off, src);
        async_copy16(bufW + r * 32 + off, W + (size_t)(tileN + r) * 576 + k0 + off);
    }
}

__global__ __launch_bounds__(128)
void conv3x3_wmma(const bf16* __restrict__ X,   // NHWC [B,H,W,64] bf16
                  const bf16* __restrict__ W,   // [N][576], k=(kh*3+kw)*64+c
                  const bf16* __restrict__ zp,  // 16B zero page
                  const float* __restrict__ bias,
                  float* __restrict__ out,      // [M,N] f32
                  int H, int Wd, int N)
{
    __shared__ __align__(16) bf16 lA[2][64 * 32];
    __shared__ __align__(16) bf16 lW[2][64 * 32];
    const int K = 576;
    int tidx = threadIdx.x, wid = tidx >> 5;
    int tileN = blockIdx.x * 64, tileM = blockIdx.y * 64;
    int HW = H * Wd;

    f32x8 acc[4];
    #pragma unroll
    for (int nt = 0; nt < 4; nt++)
        #pragma unroll
        for (int r = 0; r < 8; r++) acc[nt][r] = 0.f;

    conv_issue(X, W, zp, lA[0], lW[0], tileM, tileN, 0, H, Wd, HW, tidx);
    int cur = 0;
    for (int k0 = 0; k0 < K; k0 += 32) {
        if (k0 + 32 < K) {
            conv_issue(X, W, zp, lA[cur ^ 1], lW[cur ^ 1], tileM, tileN, k0 + 32, H, Wd, HW, tidx);
            WAIT_ASYNC_4();
        } else {
            WAIT_ASYNC_0();
        }
        __syncthreads();
        v16bf af = frag_a(lA[cur], wid * 16, 32);
        #pragma unroll
        for (int nt = 0; nt < 4; nt++) {
            v16bf bfw = frag_b(lW[cur], nt * 16, 32);
            acc[nt] = __builtin_amdgcn_wmma_f32_16x16x32_bf16(
                false, af, false, bfw, (short)0, acc[nt], false, false);
        }
        __syncthreads();
        cur ^= 1;
    }

    int lane = tidx & 31, g = lane >> 4, nl = lane & 15;
    #pragma unroll
    for (int nt = 0; nt < 4; nt++) {
        int n = tileN + nt * 16 + nl;
        float bv = bias[n];
        #pragma unroll
        for (int r = 0; r < 8; r++) {
            int m = tileM + wid * 16 + g * 8 + r;
            out[(size_t)m * N + n] = acc[nt][r] + bv;
        }
    }
}

// ---------------- 7x7 s2 stem conv, implicit GEMM (K pad 147->160) ----------
__global__ __launch_bounds__(128)
void stem_wmma(const float* __restrict__ X,     // [B,224,224,3] f32 NHWC
               const bf16* __restrict__ W,      // [64][160] bf16
               const float* __restrict__ bias,
               float* __restrict__ out)         // [B,112,112,64] f32 (== [M,64])
{
    __shared__ __align__(16) bf16 lA[64 * 32];
    __shared__ __align__(16) bf16 lW[64 * 32];
    const int K = 160;
    int tidx = threadIdx.x, wid = tidx >> 5;
    int tileM = blockIdx.x * 64;

    f32x8 acc[4];
    #pragma unroll
    for (int nt = 0; nt < 4; nt++)
        #pragma unroll
        for (int r = 0; r < 8; r++) acc[nt][r] = 0.f;

    for (int k0 = 0; k0 < K; k0 += 32) {
        for (int i = 0; i < 16; i++) {          // 2048 elements, 16 per thread
            int eid = tidx + i * 128;
            int r = eid >> 5, kk = eid & 31;
            int k = k0 + kk;
            int m = tileM + r;
            int b = m / 12544; int rem = m - b * 12544;
            int oh = rem / 112, ow = rem - (rem / 112) * 112;
            float v = 0.f;
            if (k < 147) {
                int tap = k / 3, c = k - 3 * tap;
                int kh = tap / 7, kw = tap - 7 * kh;
                int ih = oh * 2 - 3 + kh, iw = ow * 2 - 3 + kw;
                if (ih >= 0 && ih < 224 && iw >= 0 && iw < 224)
                    v = X[(((size_t)b * 224 + ih) * 224 + iw) * 3 + c];
            }
            lA[r * 32 + kk] = (bf16)v;
        }
        #pragma unroll
        for (int i = 0; i < 2; i++) {
            int cid = tidx + i * 128;
            int r = cid >> 2, off = (cid & 3) * 8;
            *(bf16x8*)&lW[r * 32 + off] = *(const bf16x8*)&W[(size_t)r * K + k0 + off];
        }
        __syncthreads();
        v16bf af = frag_a(lA, wid * 16, 32);
        #pragma unroll
        for (int nt = 0; nt < 4; nt++) {
            v16bf bfw = frag_b(lW, nt * 16, 32);
            acc[nt] = __builtin_amdgcn_wmma_f32_16x16x32_bf16(
                false, af, false, bfw, (short)0, acc[nt], false, false);
        }
        __syncthreads();
    }

    int lane = tidx & 31, g = lane >> 4, nl = lane & 15;
    #pragma unroll
    for (int nt = 0; nt < 4; nt++) {
        int n = nt * 16 + nl;
        float bv = bias[n];
        #pragma unroll
        for (int r = 0; r < 8; r++) {
            int m = tileM + wid * 16 + g * 8 + r;
            float v = acc[nt][r] + bv;
            out[(size_t)m * 64 + n] = v > 0.f ? v : 0.f;   // fused ReLU
        }
    }
}

// ---------------- weight repack / convert / zero kernels --------------------
__global__ void repack_stem_w(const float* __restrict__ w, bf16* __restrict__ o) {
    int t = blockIdx.x * blockDim.x + threadIdx.x;   // 64*160
    if (t >= 64 * 160) return;
    int oc = t / 160, k = t - oc * 160;
    float v = 0.f;
    if (k < 147) {
        int tap = k / 3, c = k - 3 * tap;
        int kh = tap / 7, kw = tap - 7 * kh;
        v = w[((oc * 3 + c) * 7 + kh) * 7 + kw];     // (64,3,7,7)
    }
    o[t] = (bf16)v;
}
__global__ void repack_conv_w(const float* __restrict__ w, bf16* __restrict__ o, int OC) {
    int t = blockIdx.x * blockDim.x + threadIdx.x;   // OC*576
    if (t >= OC * 576) return;
    int oc = t / 576, k = t - oc * 576;
    int tap = k >> 6, c = k & 63;
    int kh = tap / 3, kw = tap - 3 * kh;
    o[t] = (bf16)w[((oc * 64 + c) * 3 + kh) * 3 + kw]; // (OC,64,3,3)
}
__global__ void f32_to_bf16_k(const float* __restrict__ a, bf16* __restrict__ o, int n) {
    int t = blockIdx.x * blockDim.x + threadIdx.x;
    if (t < n) o[t] = (bf16)a[t];
}
__global__ void zero_bf16_k(bf16* __restrict__ o, int n) {
    int t = blockIdx.x * blockDim.x + threadIdx.x;
    if (t < n) o[t] = (bf16)0.f;
}

// ---------------- maxpool 2x2 s2 on NHWC; writes f32 + bf16 copies ----------
__global__ void maxpool2_dual(const float* __restrict__ in, float* __restrict__ outF,
                              bf16* __restrict__ outH, int B, int H, int W, int C) {
    int OH = H / 2, OW = W / 2;
    long long t = (long long)blockIdx.x * blockDim.x + threadIdx.x;
    long long total = (long long)B * OH * OW * C;
    if (t >= total) return;
    int c = (int)(t % C); long long rest = t / C;
    int ow = (int)(rest % OW); rest /= OW;
    int oh = (int)(rest % OH); int b = (int)(rest / OH);
    const float* p = in + (((size_t)b * H + oh * 2) * W + ow * 2) * C + c;
    size_t rs = (size_t)W * C;
    float v = fmaxf(fmaxf(p[0], p[C]), fmaxf(p[rs], p[rs + C]));
    outF[t] = v;
    outH[t] = (bf16)v;
}

// ---------------- BatchNorm (train-mode stats), deterministic ---------------
__global__ __launch_bounds__(256)
void bn_stats(const float* __restrict__ X, float* __restrict__ psum,
              float* __restrict__ psq, int C) {
    int t = threadIdx.x;
    size_t base = (size_t)blockIdx.x * 256 * C;
    for (int c = t; c < C; c += 256) {
        float s = 0.f, q = 0.f;
        for (int r = 0; r < 256; r++) {
            float v = X[base + (size_t)r * C + c];
            s += v; q += v * v;
        }
        psum[(size_t)blockIdx.x * C + c] = s;
        psq [(size_t)blockIdx.x * C + c] = q;
    }
}
__global__ void bn_finalize(const float* __restrict__ psum, const float* __restrict__ psq,
                            const float* __restrict__ gamma, const float* __restrict__ beta,
                            float* __restrict__ scale, float* __restrict__ shift,
                            int C, int nb, float invM) {
    int c = blockIdx.x * blockDim.x + threadIdx.x;
    if (c >= C) return;
    float s = 0.f, q = 0.f;
    for (int i = 0; i < nb; i++) { s += psum[(size_t)i * C + c]; q += psq[(size_t)i * C + c]; }
    float mean = s * invM;
    float var  = q * invM - mean * mean;
    float rstd = rsqrtf(var + 1e-5f);
    float sc = gamma[c] * rstd;
    scale[c] = sc;
    shift[c] = beta[c] - sc * mean;
}
__global__ void bn_silu_bf16(const float* __restrict__ X, const float* __restrict__ scale,
                             const float* __restrict__ shift, bf16* __restrict__ out,
                             long long n, int C) {
    long long t = (long long)blockIdx.x * blockDim.x + threadIdx.x;
    if (t >= n) return;
    int c = (int)(t & (C - 1));                      // C is a power of two
    float v = X[t] * scale[c] + shift[c];
    out[t] = (bf16)(v / (1.f + __expf(-v)));         // SiLU
}

// ---------------- linear recurrence s_t = 0.9 s_{t-1} + u_t -----------------
__global__ void scan_seq(const float* __restrict__ U, bf16* __restrict__ S,
                         int B, int L, int C) {
    int t = blockIdx.x * blockDim.x + threadIdx.x;
    if (t >= B * C) return;
    int b = t / C, c = t - b * C;
    const float* u = U + (size_t)b * L * C + c;
    bf16* s = S + (size_t)b * L * C + c;
    float acc = 0.f;
    for (int l = 0; l < L; l++) {
        acc = 0.9f * acc + u[(size_t)l * C];
        s[(size_t)l * C] = (bf16)acc;
    }
}

// ---------------- final FC (5 classes) with NCHW-flatten permute ------------
__global__ void fc_kernel(const float* __restrict__ P,    // NHWC [64,7,7,64]
                          const float* __restrict__ W,    // [5,3136] k=c*49+h*7+w
                          const float* __restrict__ bias,
                          float* __restrict__ out) {
    int t = blockIdx.x * blockDim.x + threadIdx.x;
    if (t >= 320) return;
    int b = t / 5, j = t - 5 * b;
    const float* p = P + (size_t)b * 3136;
    const float* w = W + (size_t)j * 3136;
    float acc = bias[j];
    for (int k = 0; k < 3136; k++) {
        int c = k / 49, hw = k - 49 * c;
        acc += w[k] * p[hw * 64 + c];
    }
    out[t] = acc;
}

// ============================================================================
extern "C" void kernel_launch(void* const* d_in, const int* in_sizes, int n_in,
                              void* d_out, int out_size, void* d_ws, size_t ws_size,
                              hipStream_t stream) {
    const float* x    = (const float*)d_in[0];
    const float* ic_w = (const float*)d_in[1];
    const float* ic_b = (const float*)d_in[2];
    const int hchA[3] = {128, 256, 512};
    const int Hs[3]   = {56, 28, 14};

    const float* bp[3][12];
    const float *fcw, *fcb;
    if (n_in >= 41) {
        for (int b = 0; b < 3; b++)
            for (int j = 0; j < 12; j++) bp[b][j] = (const float*)d_in[3 + b * 12 + j];
        fcw = (const float*)d_in[39]; fcb = (const float*)d_in[40];
    } else {  // tuple passed as one concatenated blob per block
        for (int b = 0; b < 3; b++) {
            const float* base = (const float*)d_in[3 + b];
            size_t h = (size_t)hchA[b];
            size_t sz[12] = {h * 64 * 9, h, h, h, h * h, h, h * h, h, h * h, h, 64 * h, 64};
            size_t off = 0;
            for (int j = 0; j < 12; j++) { bp[b][j] = base + off; off += sz[j]; }
        }
        fcw = (const float*)d_in[6]; fcb = (const float*)d_in[7];
    }

    // ---- workspace layout (ping-pong regions; peak ~495 MB) ----
    char* ws = (char*)d_ws;
    float* STEM = (float*)ws;                          // 205.5MB stem out; later GATE
    float* GATE = STEM;
    size_t O_AIN = 205520896;                          // 64*112*112*64*4
    float* AIN  = (float*)(ws + O_AIN);                // 51.4MB block in (f32, skip)
    size_t O_C  = O_AIN + 51380224;
    float* CBUF = (float*)(ws + O_C);                  // 102.8MB conv out / U / blk out
    size_t O_XF = O_C + 102760448;
    bf16*  XF   = (bf16*)(ws + O_XF);                  // 51.4MB xf bf16, reused for s
    size_t O_Y1 = O_XF + 51380224;
    bf16*  Y1   = (bf16*)(ws + O_Y1);                  // 51.4MB
    size_t O_W  = O_Y1 + 51380224;
    bf16*  WCONV = (bf16*)(ws + O_W);                  // <=512*576
    bf16*  WG = WCONV + 512 * 576;
    bf16*  WI = WG + 512 * 512;
    bf16*  WSm = WI + 512 * 512;
    bf16*  WO = WSm + 512 * 512;
    bf16*  WSTEM = WO + 64 * 512;
    float* SCALE = (float*)(WSTEM + 64 * 160);
    float* SHIFT = SCALE + 512;
    float* PSUM  = SHIFT + 512;                        // <=784*512 partials
    float* PSQ   = PSUM + 784 * 512;
    bf16*  AINH  = (bf16*)(PSQ + 784 * 512);           // 25.7MB bf16 activation copy
    bf16*  ZP    = AINH + 64 * 56 * 56 * 64;           // 128-elem zero page

    // ---- zero page for conv halo DMA ----
    zero_bf16_k<<<1, 128, 0, stream>>>(ZP, 128);

    // ---- stem: conv7x7 s2 + relu (WMMA) then maxpool ----
    repack_stem_w<<<(64 * 160 + 255) / 256, 256, 0, stream>>>(ic_w, WSTEM);
    stem_wmma<<<12544, 128, 0, stream>>>(x, WSTEM, ic_b, STEM);   // M=64*112*112
    {
        long long tot = 64LL * 56 * 56 * 64;
        maxpool2_dual<<<(unsigned)((tot + 255) / 256), 256, 0, stream>>>(STEM, AIN, AINH,
                                                                         64, 112, 112, 64);
    }

    // ---- three mamba blocks ----
    for (int b = 0; b < 3; b++) {
        int hch = hchA[b], H = Hs[b], Wd = H;
        int L = H * Wd, M = 64 * L;
        const float *cw = bp[b][0], *cb = bp[b][1], *gamma = bp[b][2], *beta = bp[b][3];
        const float *gw = bp[b][4], *gb = bp[b][5], *iw = bp[b][6], *ib = bp[b][7];
        const float *sw = bp[b][8], *sb = bp[b][9], *ow = bp[b][10], *ob = bp[b][11];

        repack_conv_w<<<(hch * 576 + 255) / 256, 256, 0, stream>>>(cw, WCONV, hch);
        f32_to_bf16_k<<<(hch * hch + 255) / 256, 256, 0, stream>>>(gw, WG, hch * hch);
        f32_to_bf16_k<<<(hch * hch + 255) / 256, 256, 0, stream>>>(iw, WI, hch * hch);
        f32_to_bf16_k<<<(hch * hch + 255) / 256, 256, 0, stream>>>(sw, WSm, hch * hch);
        f32_to_bf16_k<<<(64 * hch + 255) / 256, 256, 0, stream>>>(ow, WO, 64 * hch);

        // conv3x3 + bias  -> CBUF [M,hch]   (async implicit GEMM on bf16 copy)
        conv3x3_wmma<<<dim3(hch / 64, M / 64), 128, 0, stream>>>(AINH, WCONV, ZP, cb, CBUF,
                                                                 H, Wd, hch);

        // train-mode BN stats + silu -> XF bf16
        int nb = M / 256;
        bn_stats<<<nb, 256, 0, stream>>>(CBUF, PSUM, PSQ, hch);
        bn_finalize<<<(hch + 63) / 64, 64, 0, stream>>>(PSUM, PSQ, gamma, beta, SCALE, SHIFT,
                                                        hch, nb, 1.f / (float)M);
        long long n = (long long)M * hch;
        bn_silu_bf16<<<(unsigned)((n + 255) / 256), 256, 0, stream>>>(CBUF, SCALE, SHIFT, XF, n, hch);

        // gate = sigmoid(xf@gw^T+gb) -> GATE ; u = xf@iw^T+ib -> CBUF (conv buf dead)
        gemm_nt_bf16<<<dim3(hch / 64, M / 64), 128, 0, stream>>>(
            XF, WG, gb, nullptr, GATE, nullptr, M, hch, hch, EPI_SIGMOID);
        gemm_nt_bf16<<<dim3(hch / 64, M / 64), 128, 0, stream>>>(
            XF, WI, ib, nullptr, CBUF, nullptr, M, hch, hch, EPI_BIAS_F32);

        // s_t = 0.9 s_{t-1} + u_t  (XF region dead -> holds s as bf16)
        scan_seq<<<(64 * hch + 255) / 256, 256, 0, stream>>>(CBUF, XF, 64, L, hch);

        // y1 = gate * (s@sw^T+sb) -> Y1 bf16 ; y2 = y1@ow^T+ob + skip -> CBUF [M,64]
        gemm_nt_bf16<<<dim3(hch / 64, M / 64), 128, 0, stream>>>(
            XF, WSm, sb, GATE, nullptr, Y1, M, hch, hch, EPI_GATE_BF16);
        gemm_nt_bf16<<<dim3(1, M / 64), 128, 0, stream>>>(
            Y1, WO, ob, AIN, CBUF, nullptr, M, 64, hch, EPI_SKIP_F32);

        // maxpool -> AIN/AINH become next block's input (skip is dead now)
        long long tot = 64LL * (H / 2) * (Wd / 2) * 64;
        maxpool2_dual<<<(unsigned)((tot + 255) / 256), 256, 0, stream>>>(CBUF, AIN, AINH,
                                                                         64, H, Wd, 64);
    }

    // ---- final FC: [64,3136]@fc_w^T + fc_b -> d_out [64,5] ----
    fc_kernel<<<5, 64, 0, stream>>>(AIN, fcw, fcb, (float*)d_out);
}
